// Matcher_43190191128721
// MI455X (gfx1250) — matrix operand order repaired and checked by reading, not compiled, not run on hardware
//
#include <hip/hip_runtime.h>
#include <hip/hip_bf16.h>

// ---------------------------------------------------------------------------
// GCN matcher restructured as:
//   prep (1 block):  E1 = emb@W1  [129x128 f32]
//                    class branch -> G[c][s][k] = (M^T o_c W2^T)   [32x640 f16, pad]
//                    const[c] = b2 . (sum_t o_c[t])               [32 f32]
//   main (512 blocks x 256 thr): per 16-pixel tile:
//        r_p = relu(M (E1 gathers) + b1)  -> f16 LDS stage (vectorized, per-wave)
//        sim[p, 0..20] = WMMA( r[16x640] , G^T[640x32] ) + const  (v_wmma_f32_16x16x32_f16)
// ---------------------------------------------------------------------------

typedef _Float16 v16h __attribute__((ext_vector_type(16)));
typedef _Float16 h8   __attribute__((ext_vector_type(8)));
typedef _Float16 h4   __attribute__((ext_vector_type(4)));
typedef float    v8f  __attribute__((ext_vector_type(8)));
typedef float    v4f  __attribute__((ext_vector_type(4)));

#define IS2 0.70710678118654752f

__device__ __forceinline__ h4 relu_cvt_h4(v4f v) {
    h4 r;
    r[0] = (_Float16)fmaxf(v[0], 0.f);
    r[1] = (_Float16)fmaxf(v[1], 0.f);
    r[2] = (_Float16)fmaxf(v[2], 0.f);
    r[3] = (_Float16)fmaxf(v[3], 0.f);
    return r;
}

// ------------------------------- prep kernel -------------------------------
__global__ __launch_bounds__(256) void gcn_prep(
    const int* __restrict__ class_nodes,   // [21*5]
    const float* __restrict__ emb,         // [129*21]
    const float* __restrict__ W1,          // [21*128]
    const float* __restrict__ b1,          // [128]
    const float* __restrict__ W2,          // [128*21]
    const float* __restrict__ b2,          // [21]
    float* __restrict__ E1g,               // [129*128]
    _Float16* __restrict__ Gg,             // [32*640]
    float* __restrict__ Cg)                // [32]
{
    __shared__ float sE1[129 * 128];       // 66048 B
    __shared__ float sH [21 * 5 * 128];    // 53760 B
    __shared__ float sQ [21 * 5 * 21];     //  8820 B
    __shared__ float sO [21 * 5 * 21];     //  8820 B
    const int t = threadIdx.x;

    // E1 = emb @ W1
    for (int i = t; i < 129 * 128; i += 256) {
        int row = i >> 7, col = i & 127;
        float acc = 0.f;
        for (int d = 0; d < 21; ++d) acc += emb[row * 21 + d] * W1[d * 128 + col];
        sE1[i] = acc;
        E1g[i] = acc;
    }
    __syncthreads();

    // h = relu(M (x W1) + b1) for the 21 class graphs, via E1 gathers
    for (int i = t; i < 21 * 5 * 128; i += 256) {
        int j = i & 127, cs = i >> 7, s = cs % 5, c = cs / 5;
        int id  = class_nodes[c * 5 + s];
        int id0 = class_nodes[c * 5 + 0];
        float v = (s == 0) ? (sE1[id * 128 + j] + b1[j])
                           : (0.5f * sE1[id * 128 + j] + IS2 * sE1[id0 * 128 + j] + b1[j]);
        sH[i] = fmaxf(v, 0.f);
    }
    __syncthreads();

    // q = h @ W2
    for (int i = t; i < 21 * 5 * 21; i += 256) {
        int d = i % 21, cs = i / 21;
        const float* hrow = &sH[cs * 128];
        float acc = 0.f;
        for (int k = 0; k < 128; ++k) acc += hrow[k] * W2[k * 21 + d];
        sQ[i] = acc;
    }
    __syncthreads();

    // o = M q + b2
    for (int i = t; i < 21 * 5 * 21; i += 256) {
        int d = i % 21, cs = i / 21, s = cs % 5, c = cs / 5;
        float v = (s == 0) ? sQ[(c * 5) * 21 + d]
                           : 0.5f * sQ[cs * 21 + d] + IS2 * sQ[(c * 5) * 21 + d];
        sO[i] = v + b2[d];
    }
    __syncthreads();

    // G[c][s][k] = sum_d W2[k][d] * (M^T o_c)[s][d]   (pad classes 21..31 with 0)
    for (int i = t; i < 32 * 5 * 128; i += 256) {
        int k = i & 127, cs = i >> 7, s = cs % 5, c = cs / 5;
        float acc = 0.f;
        if (c < 21) {
            for (int d = 0; d < 21; ++d) {
                float P;
                if (s == 0)
                    P = sO[(c * 5 + 0) * 21 + d] +
                        IS2 * (sO[(c * 5 + 1) * 21 + d] + sO[(c * 5 + 2) * 21 + d] +
                               sO[(c * 5 + 3) * 21 + d] + sO[(c * 5 + 4) * 21 + d]);
                else
                    P = 0.5f * sO[(c * 5 + s) * 21 + d];
                acc += W2[k * 21 + d] * P;
            }
        }
        Gg[c * 640 + s * 128 + k] = (_Float16)acc;
    }

    // const[c] = sum_t sum_d b2[d] * o_c[t][d]
    for (int c = t; c < 32; c += 256) {
        float acc = 0.f;
        if (c < 21)
            for (int tt = 0; tt < 5; ++tt)
                for (int d = 0; d < 21; ++d) acc += b2[d] * sO[(c * 5 + tt) * 21 + d];
        Cg[c] = acc;
    }
}

// ------------------------------- main kernel -------------------------------
__global__ __launch_bounds__(256) void gcn_main(
    const int* __restrict__ nodes,       // [pixels*5]
    const float* __restrict__ E1g,       // [129*128]
    const _Float16* __restrict__ Gg,     // [32*640]
    const float* __restrict__ Cg,        // [32]
    const float* __restrict__ b1,        // [128]
    float* __restrict__ out)             // [pixels*21]
{
    __shared__ __align__(16) float sE1[129 * 128];         //  66048 B
    __shared__ __align__(16) _Float16 sG[32 * 640];        //  40960 B
    __shared__ float sC[32];
    __shared__ __align__(16) float sB1[128];
    __shared__ __align__(16) _Float16 sR[8][16 * 640];     // 163840 B (~266 KB total)

    const int t    = threadIdx.x;
    const int lane = t & 31;
    // force wave index into an SGPR so per-pixel node-id loads become s_load
    const int wv   = __builtin_amdgcn_readfirstlane(t >> 5);

    // cooperative fill of E1 / G / const / b1 (one time, vectorized)
    {
        const v4f* E1v = (const v4f*)E1g;
        v4f* sE1v = (v4f*)sE1;
        for (int i = t; i < (129 * 128) / 4; i += 256) sE1v[i] = E1v[i];
        const uint4* Gv = (const uint4*)Gg;
        uint4* sGv = (uint4*)sG;
        for (int i = t; i < (32 * 640) / 8; i += 256) sGv[i] = Gv[i];
        if (t < 32)  sC[t]  = Cg[t];
        if (t < 128) sB1[t] = b1[t];
    }
    __syncthreads();

    // hoisted per-lane constants
    const int   j4 = lane << 2;                       // this lane's 4 hidden dims
    const v4f   bb = *(const v4f*)&sB1[j4];
    _Float16*   rbase = &sR[wv][0];

    const int mrow = lane & 15;
    const int aoff = (lane < 16) ? 0 : 16;   // A frag: lanes<16 K 0..15; else K 16..31
    const int boff = (lane < 16) ? 0 : 16;   // B frag: lanes<16 K 0..15; else K 16..31
    const _Float16* rrow = &sR[wv][mrow * 640];
    const _Float16* g0   = &sG[(lane & 15) * 640];
    const _Float16* g1   = &sG[(16 + (lane & 15)) * 640];
    const int   rb  = (lane < 16) ? 0 : 8;
    const int   col = lane & 15;
    const float c0  = sC[col];
    const float c1  = sC[16 + col];

    // 4 tiles per wave, 8 waves per block, 16 pixels per tile -> 512 pixels/block
    for (int it = 0; it < 4; ++it) {
        const int pbase = (blockIdx.x * 32 + it * 8 + wv) << 4;
        const int* __restrict__ np = nodes + pbase * 5;   // uniform -> scalar loads

        // ---- stage r = relu(M (xW1) + b1) as f16, row-major [pixel][s*128+j] ----
        for (int p = 0; p < 16; ++p) {
            const int id0 = np[p * 5];
            const v4f e0  = *(const v4f*)&sE1[id0 * 128 + j4];
            const v4f t0  = e0 * IS2 + bb;               // shared term for s=1..4
            _Float16* dst = rbase + p * 640 + j4;
            *(h4*)dst = relu_cvt_h4(e0 + bb);            // s = 0
#pragma unroll
            for (int s = 1; s < 5; ++s) {
                const int id = np[p * 5 + s];
                const v4f e  = *(const v4f*)&sE1[id * 128 + j4];
                *(h4*)(dst + s * 128) = relu_cvt_h4(e * 0.5f + t0);
            }
        }
        // wave-private sR: only need this wave's LDS stores complete
        asm volatile("s_wait_dscnt 0" ::: "memory");

        // ---- GEMM: [16 x 640] x [640 x 32] via v_wmma_f32_16x16x32_f16 ----
        v8f acc0 = {0.f, 0.f, 0.f, 0.f, 0.f, 0.f, 0.f, 0.f};
        v8f acc1 = acc0;
#pragma unroll
        for (int kb = 0; kb < 640; kb += 32) {
            union { v16h v; h8 h[2]; } A, B0, B1;
            A.h[0]  = *(const h8*)(rrow + kb + aoff);
            A.h[1]  = *(const h8*)(rrow + kb + aoff + 8);
            B0.h[0] = *(const h8*)(g0 + kb + boff);
            B0.h[1] = *(const h8*)(g0 + kb + boff + 8);
            B1.h[0] = *(const h8*)(g1 + kb + boff);
            B1.h[1] = *(const h8*)(g1 + kb + boff + 8);
            acc0 = __builtin_amdgcn_wmma_f32_16x16x32_f16(false, A.v, false, B0.v,
                                                          (short)0, acc0, false, false);
            acc1 = __builtin_amdgcn_wmma_f32_16x16x32_f16(false, A.v, false, B1.v,
                                                          (short)0, acc1, false, false);
        }

        // ---- epilogue: C layout -> VGPR i = row (i + 8*(lane>=16)), col = lane&15 ----
#pragma unroll
        for (int i = 0; i < 8; ++i) {
            int p = pbase + rb + i;
            out[p * 21 + col] = acc0[i] + c0;
            if (col < 5) out[p * 21 + 16 + col] = acc1[i] + c1;
        }
        // all GEMM ds reads of sR done before next iteration's stores
        asm volatile("s_wait_dscnt 0" ::: "memory");
    }
}

// ------------------------------- launcher ----------------------------------
extern "C" void kernel_launch(void* const* d_in, const int* in_sizes, int n_in,
                              void* d_out, int out_size, void* d_ws, size_t ws_size,
                              hipStream_t stream) {
    const int*   inst = (const int*)d_in[0];     // [B*H*W*5]
    const int*   cls  = (const int*)d_in[1];     // [21*5]
    const float* emb  = (const float*)d_in[2];   // [129*21]
    const float* W1   = (const float*)d_in[3];   // [21*128]
    const float* b1   = (const float*)d_in[4];   // [128]
    const float* W2   = (const float*)d_in[5];   // [128*21]
    const float* b2   = (const float*)d_in[6];   // [21]
    float* out = (float*)d_out;

    // workspace: E1 f32 (66048B) | G f16 (40960B) | const f32 (128B)
    float*    E1g = (float*)d_ws;
    _Float16* Gg  = (_Float16*)((char*)d_ws + 66048);
    float*    Cg  = (float*)((char*)d_ws + 66048 + 40960);

    gcn_prep<<<1, 256, 0, stream>>>(cls, emb, W1, b1, W2, b2, E1g, Gg, Cg);

    const int pixels = in_sizes[0] / 5;          // 262144
    const int grid   = pixels / (16 * 32);       // 512 blocks, 32 tiles each
    gcn_main<<<grid, 256, 0, stream>>>(inst, E1g, Gg, Cg, b1, out);
}
